// VGAE_encoder_13640816132536
// MI455X (gfx1250) — matrix operand order repaired and checked by reading, not compiled, not run on hardware
//
#include <hip/hip_runtime.h>
#include <hip/hip_bf16.h>
#include <cstdint>
#include <cstddef>

// ---------------------------------------------------------------------------
// VGAE encoder (NNConv x4) fused for MI455X / gfx1250, wave32 + WMMA f16.
//   N=50000 nodes, E=150000 edges, IN=16, HID=32, LAT=16, EF=8, HE=128
// Strategy: fuse edge-MLP GEMMs + per-edge matvec + scatter so the huge
// per-edge weight tensor W [E, din*dout] never touches HBM (~2.4GB avoided).
// w2/b2 staged to LDS via global_load_async_to_lds_b128; w1 B-fragments and
// b1 are hoisted into persistent VGPRs (loop-invariant per wave).
// ---------------------------------------------------------------------------

#define NN    50000
#define EE    150000
#define IN_F  16
#define HID_F 32
#define LAT_F 16
#define EF_F  8
#define HE_F  128

typedef __attribute__((ext_vector_type(16))) _Float16 v16h;
typedef __attribute__((ext_vector_type(8)))  _Float16 v8h;
typedef __attribute__((ext_vector_type(8)))  float    v8f;

// ---- WMMA fragment index helpers (per CDNA5 ISA 7.12.2 layouts) -----------
// A (16x32 f16): lane group g=lane>>4; K = g*8 + 2v+h + (v>=4 ? 8 : 0), M=lane&15
__device__ __forceinline__ int aK(int lane, int v, int h) {
  return ((lane >> 4) << 3) + 2 * v + h + (v >= 4 ? 8 : 0);
}
// B (32x16 f16): K = g*8 + 2v+h + (v>=4 ? 8 : 0), N = lane&15
__device__ __forceinline__ int bK(int lane, int v, int h) {
  return ((lane >> 4) << 3) + 2 * v + h + (v >= 4 ? 8 : 0);
}

// ---- cooperative global->LDS async copy (16B granules, ASYNCcnt) ----------
__device__ __forceinline__ void async_copy_lds(void* lds, const void* gmem,
                                               int bytes, int tid, int nthr) {
  unsigned lbase = (unsigned)(uintptr_t)lds;
  for (int i = tid * 16; i < bytes; i += nthr * 16) {
    unsigned l = lbase + i;
    const char* g = (const char*)gmem + i;
    asm volatile("global_load_async_to_lds_b128 %0, %1, off"
                 :: "v"(l), "v"(g) : "memory");
  }
}

// ---- prep: stage edge_attr into A-fragment tiles (zero-padded K 8->32) ----
__global__ void stage_ea_kernel(const float* __restrict__ ea,
                                _Float16* __restrict__ eaA, int ntiles) {
  int t = blockIdx.x * blockDim.x + threadIdx.x;
  if (t >= ntiles * 32) return;
  int tile = t >> 5, lane = t & 31;
  int e = tile * 16 + (lane & 15);
  _Float16* o = eaA + (size_t)tile * 512 + lane * 16;
#pragma unroll
  for (int v = 0; v < 8; ++v)
#pragma unroll
    for (int h = 0; h < 2; ++h) {
      int K = aK(lane, v, h);
      o[2 * v + h] = (K < EF_F) ? (_Float16)ea[(size_t)e * EF_F + K]
                                : (_Float16)0.f;
    }
}

// ---- prep: stage w1 [8,128] into 8 B-fragments (zero-padded K 8->32) ------
__global__ void stage_w1_kernel(const float* __restrict__ w1,
                                _Float16* __restrict__ w1B) {
  int t = blockIdx.x * blockDim.x + threadIdx.x;
  if (t >= 8 * 32) return;
  int nt = t >> 5, lane = t & 31;
  _Float16* o = w1B + nt * 512 + lane * 16;
  int col = nt * 16 + (lane & 15);
#pragma unroll
  for (int v = 0; v < 8; ++v)
#pragma unroll
    for (int h = 0; h < 2; ++h) {
      int K = bK(lane, v, h);
      o[2 * v + h] = (K < EF_F) ? (_Float16)w1[(size_t)K * HE_F + col]
                                : (_Float16)0.f;
    }
}

// ---- prep: stage w2 [128, NT*16] into [ks=4][NT] B-fragments --------------
__global__ void stage_w2_kernel(const float* __restrict__ w2,
                                _Float16* __restrict__ w2B, int NT) {
  int t = blockIdx.x * blockDim.x + threadIdx.x;
  if (t >= 4 * NT * 32) return;
  int lane = t & 31;
  int nt = (t >> 5) % NT;
  int ks = (t >> 5) / NT;
  _Float16* o = w2B + ((size_t)(ks * NT + nt)) * 512 + lane * 16;
  int col = nt * 16 + (lane & 15);
  int ncol = NT * 16;
#pragma unroll
  for (int v = 0; v < 8; ++v)
#pragma unroll
    for (int h = 0; h < 2; ++h) {
      int K = ks * 32 + bK(lane, v, h);
      o[2 * v + h] = (_Float16)w2[(size_t)K * ncol + col];
    }
}

// ---- misc small kernels ---------------------------------------------------
__global__ void zero_kernel(float* p, int n) {
  int t = blockIdx.x * blockDim.x + threadIdx.x;
  if (t < n) p[t] = 0.f;
}

__global__ void edge_count_kernel(const long long* __restrict__ dst,
                                  float* __restrict__ cnt, int e) {
  int t = blockIdx.x * blockDim.x + threadIdx.x;
  if (t < e) atomicAdd(cnt + (int)dst[t], 1.f);
}

// ---- fused NNConv edge kernel --------------------------------------------
// One wave = one tile of 16 edges. Per tile:
//   h[16,128] = relu(eaA @ w1F + b1)          (8 WMMAs, w1F in registers)
//   for each 16-col slice of W = h@w2+b2:     (4 WMMAs each, NT slices)
//       msg[e, o] += x_src[e, i] * W[e, i, o] (VALU on f32 accumulators)
//   scatter-add msg to agg[dst]               (global f32 atomics)
// w2 fragments + b2 live in LDS (async-staged once per workgroup).
template <int DIN, int DOUT>
__global__ __launch_bounds__(256) void nnconv_edge_kernel(
    const float* __restrict__ x, const long long* __restrict__ src,
    const long long* __restrict__ dst, const _Float16* __restrict__ eaA,
    const _Float16* __restrict__ w1B, const float* __restrict__ b1,
    const _Float16* __restrict__ w2B, const float* __restrict__ b2,
    float* __restrict__ agg, int ntiles) {
  constexpr int NT  = (DIN * DOUT) / 16;
  constexpr int NCH = DOUT / 16;
  constexpr int W2_BYTES = 4 * NT * 512 * 2;
  constexpr int B2_BYTES = NT * 16 * 4;
  constexpr int WAVE_BYTES = 16 * HE_F * 2 + DIN * 16 * 4 + 16 * 4;

  extern __shared__ char smem[];
  _Float16* w2l = (_Float16*)smem;
  float*    b2l = (float*)(smem + W2_BYTES);
  char* pw = smem + W2_BYTES + B2_BYTES;

  const int wib  = threadIdx.x >> 5;
  const int lane = threadIdx.x & 31;
  _Float16* h_plain = (_Float16*)(pw + (size_t)wib * WAVE_BYTES);
  float*    x_t     = (float*)(pw + (size_t)wib * WAVE_BYTES + 16 * HE_F * 2);
  int*      dstl    = (int*)(pw + (size_t)wib * WAVE_BYTES + 16 * HE_F * 2 +
                             DIN * 16 * 4);

  // async-stage streamed weight operands into LDS (ASYNCcnt tracked)
  async_copy_lds(w2l, w2B, W2_BYTES, threadIdx.x, blockDim.x);
  async_copy_lds(b2l, b2, B2_BYTES, threadIdx.x, blockDim.x);

  const int g  = lane >> 4;
  const int Nc = lane & 15;

  // hoist loop-invariant w1 B-fragments + b1 into persistent registers
  v16h w1F[8];
  float b1v[8];
#pragma unroll
  for (int nt = 0; nt < 8; ++nt) {
    w1F[nt] = *(const v16h*)(w1B + nt * 512 + lane * 16);
    b1v[nt] = b1[nt * 16 + Nc];
  }

  asm volatile("s_wait_asynccnt 0" ::: "memory");
  __syncthreads();

  const int nwaves = gridDim.x * (blockDim.x >> 5);
  const int wave0  = blockIdx.x * (blockDim.x >> 5) + wib;

  for (int tile = wave0; tile < ntiles; tile += nwaves) {
    if (tile + nwaves < ntiles)
      __builtin_prefetch(eaA + (size_t)(tile + nwaves) * 512, 0, 1);

    v16h aE = *(const v16h*)(eaA + (size_t)tile * 512 + lane * 16);

    // stage dst indices + gathered x[src] rows, TRANSPOSED: x_t[i][edge]
    {
      long long e0 = (long long)tile * 16 + Nc;
      if (lane < 16) dstl[Nc] = (int)dst[e0];
      int srow = (int)src[e0];
      const float* xs = x + (size_t)srow * DIN + g * (DIN / 2);
      float r[DIN / 2];
#pragma unroll
      for (int j = 0; j < DIN / 2; j += 4)
        *(float4*)(r + j) = *(const float4*)(xs + j);
#pragma unroll
      for (int j = 0; j < DIN / 2; ++j)
        x_t[(g * (DIN / 2) + j) * 16 + Nc] = r[j];
    }
    __builtin_amdgcn_wave_barrier();

    // h = relu(ea@w1 + b1) -> LDS row-major f16 [16][128]
    // (independent accumulators, register-resident B operands: pipelines)
    v8f hacc[8];
#pragma unroll
    for (int nt = 0; nt < 8; ++nt) {
      v8f z = {};
      hacc[nt] = __builtin_amdgcn_wmma_f32_16x16x32_f16(
          false, aE, false, w1F[nt], (short)0, z, false, false);
    }
#pragma unroll
    for (int nt = 0; nt < 8; ++nt) {
#pragma unroll
      for (int v = 0; v < 8; ++v) {
        float hv = hacc[nt][v] + b1v[nt];
        hv = hv > 0.f ? hv : 0.f;
        h_plain[(v + 8 * g) * HE_F + nt * 16 + Nc] = (_Float16)hv;
      }
    }
    __builtin_amdgcn_wave_barrier();

    // rebuild h as 4 A-fragments (16x32 each); per-lane K set is two
    // contiguous 8-half runs -> two 16B LDS loads per fragment
    v16h hA[4];
#pragma unroll
    for (int f = 0; f < 4; ++f) {
      v8h lo = *(const v8h*)(h_plain + Nc * HE_F + f * 32 + g * 8);
      v8h hi = *(const v8h*)(h_plain + Nc * HE_F + f * 32 + 16 + g * 8);
      hA[f] = __builtin_shufflevector(lo, hi, 0, 1, 2, 3, 4, 5, 6, 7, 8, 9, 10,
                                      11, 12, 13, 14, 15);
    }

    int dstv[8];
#pragma unroll
    for (int v = 0; v < 8; ++v) dstv[v] = dstl[v + 8 * g];

    v8f macc[NCH];
#pragma unroll
    for (int c = 0; c < NCH; ++c)
#pragma unroll
      for (int v = 0; v < 8; ++v) macc[c][v] = 0.f;

    // W-GEMM fused with per-edge contraction, 16 output cols at a time.
    // io = nt*16 + N -> i = io/DOUT, o = io%DOUT  (16 divides DOUT)
#pragma unroll 2
    for (int nt0 = 0; nt0 < NT; nt0 += NCH) {
      const int i = nt0 / NCH;
      float xcol[8];
      *(float4*)(xcol)     = *(const float4*)(x_t + i * 16 + 8 * g);
      *(float4*)(xcol + 4) = *(const float4*)(x_t + i * 16 + 8 * g + 4);
#pragma unroll
      for (int c = 0; c < NCH; ++c) {
        const int nt = nt0 + c;
        v8f acc = {};
#pragma unroll
        for (int ks = 0; ks < 4; ++ks) {
          v16h bF =
              *(const v16h*)(w2l + (size_t)(ks * NT + nt) * 512 + lane * 16);
          acc = __builtin_amdgcn_wmma_f32_16x16x32_f16(
              false, hA[ks], false, bF, (short)0, acc, false, false);
        }
        float b2v = b2l[nt * 16 + Nc];
#pragma unroll
        for (int v = 0; v < 8; ++v)
          macc[c][v] += xcol[v] * (acc[v] + b2v);
      }
    }

    // scatter msg to agg[dst]
#pragma unroll
    for (int c = 0; c < NCH; ++c)
#pragma unroll
      for (int v = 0; v < 8; ++v)
        atomicAdd(agg + (size_t)dstv[v] * DOUT + c * 16 + Nc, macc[c][v]);
  }
}

// ---- per-node epilogue: out = act(agg[/cnt] + x@root + bias) --------------
template <int DIN, int DOUT, bool MEAN, bool RELU>
__global__ void node_finalize_kernel(const float* __restrict__ xin,
                                     const float* __restrict__ agg,
                                     const float* __restrict__ cnt,
                                     const float* __restrict__ root,
                                     const float* __restrict__ bias,
                                     float* __restrict__ out, int n) {
  int t = blockIdx.x * blockDim.x + threadIdx.x;
  if (t >= n * DOUT) return;
  int node = t / DOUT, o = t % DOUT;
  float v = agg[t];
  if (MEAN) {
    float c = cnt[node];
    v /= (c > 1.f ? c : 1.f);
  }
  float s = bias[o];
#pragma unroll
  for (int i = 0; i < DIN; ++i)
    s += xin[(size_t)node * DIN + i] * root[i * DOUT + o];
  v += s;
  if (RELU) v = v > 0.f ? v : 0.f;
  out[t] = v;
}

// ---------------------------------------------------------------------------
extern "C" void kernel_launch(void* const* d_in, const int* in_sizes, int n_in,
                              void* d_out, int out_size, void* d_ws,
                              size_t ws_size, hipStream_t stream) {
  (void)in_sizes; (void)n_in; (void)out_size; (void)ws_size;
  const float* x = (const float*)d_in[0];
  const long long* ei = (const long long*)d_in[1];
  const float* ea = (const float*)d_in[2];
  const long long* src = ei;
  const long long* dst = ei + EE;
  // per-conv params: base = 3 + 6*c : w1,b1,w2,b2,root,bias
  const float* P[24];
  for (int i = 0; i < 24; ++i) P[i] = (const float*)d_in[3 + i];

  const int ntiles = EE / 16;  // 9375
  const int NTs[4] = {(IN_F * HID_F) / 16, (HID_F * HID_F) / 16,
                      (HID_F * LAT_F) / 16, (HID_F * LAT_F) / 16};

  // ---- workspace carve ----
  char* w = (char*)d_ws;
  auto carve = [&](size_t bytes) {
    char* p = w;
    w += (bytes + 255) & ~(size_t)255;
    return p;
  };
  _Float16* eaA = (_Float16*)carve((size_t)ntiles * 512 * 2);
  _Float16* w1B[4];
  _Float16* w2B[4];
  for (int c = 0; c < 4; ++c) w1B[c] = (_Float16*)carve(8 * 512 * 2);
  for (int c = 0; c < 4; ++c)
    w2B[c] = (_Float16*)carve((size_t)4 * NTs[c] * 512 * 2);
  float* cnt = (float*)carve((size_t)NN * 4);
  float* agg = (float*)carve((size_t)NN * HID_F * 4);
  float* h1  = (float*)carve((size_t)NN * HID_F * 4);
  float* h2  = (float*)carve((size_t)NN * HID_F * 4);

  // ---- prep: counts + fragment staging ----
  zero_kernel<<<(NN + 255) / 256, 256, 0, stream>>>(cnt, NN);
  edge_count_kernel<<<(EE + 255) / 256, 256, 0, stream>>>(dst, cnt, EE);
  stage_ea_kernel<<<(ntiles * 32 + 255) / 256, 256, 0, stream>>>(ea, eaA,
                                                                 ntiles);
  for (int c = 0; c < 4; ++c) {
    stage_w1_kernel<<<1, 256, 0, stream>>>(P[6 * c + 0], w1B[c]);
    int th = 4 * NTs[c] * 32;
    stage_w2_kernel<<<(th + 255) / 256, 256, 0, stream>>>(P[6 * c + 2], w2B[c],
                                                          NTs[c]);
  }

  auto lds_sz = [&](int NT, int DIN) {
    return (size_t)4 * NT * 512 * 2       // w2 fragments
           + (size_t)NT * 16 * 4          // b2
           + 8 * ((size_t)16 * HE_F * 2 + (size_t)DIN * 16 * 4 + 16 * 4);
  };
  const int CONV_BLOCKS = 128;

  // ---- conv1: 16 -> 32, mean, relu ----
  zero_kernel<<<(NN * HID_F + 255) / 256, 256, 0, stream>>>(agg, NN * HID_F);
  nnconv_edge_kernel<16, 32>
      <<<CONV_BLOCKS, 256, lds_sz(NTs[0], 16), stream>>>(
          x, src, dst, eaA, w1B[0], P[1], w2B[0], P[3], agg, ntiles);
  node_finalize_kernel<16, 32, true, true>
      <<<(NN * HID_F + 255) / 256, 256, 0, stream>>>(x, agg, cnt, P[4], P[5],
                                                     h1, NN);

  // ---- conv2: 32 -> 32, mean, relu ----
  zero_kernel<<<(NN * HID_F + 255) / 256, 256, 0, stream>>>(agg, NN * HID_F);
  nnconv_edge_kernel<32, 32>
      <<<CONV_BLOCKS, 256, lds_sz(NTs[1], 32), stream>>>(
          h1, src, dst, eaA, w1B[1], P[7], w2B[1], P[9], agg, ntiles);
  node_finalize_kernel<32, 32, true, true>
      <<<(NN * HID_F + 255) / 256, 256, 0, stream>>>(h1, agg, cnt, P[10],
                                                     P[11], h2, NN);

  float* mu = (float*)d_out;
  float* lv = (float*)d_out + (size_t)NN * LAT_F;

  // ---- conv_mu: 32 -> 16, sum, linear ----
  zero_kernel<<<(NN * LAT_F + 255) / 256, 256, 0, stream>>>(agg, NN * LAT_F);
  nnconv_edge_kernel<32, 16>
      <<<CONV_BLOCKS, 256, lds_sz(NTs[2], 32), stream>>>(
          h2, src, dst, eaA, w1B[2], P[13], w2B[2], P[15], agg, ntiles);
  node_finalize_kernel<32, 16, false, false>
      <<<(NN * LAT_F + 255) / 256, 256, 0, stream>>>(h2, agg, cnt, P[16],
                                                     P[17], mu, NN);

  // ---- conv_logvar: 32 -> 16, sum, linear ----
  zero_kernel<<<(NN * LAT_F + 255) / 256, 256, 0, stream>>>(agg, NN * LAT_F);
  nnconv_edge_kernel<32, 16>
      <<<CONV_BLOCKS, 256, lds_sz(NTs[3], 32), stream>>>(
          h2, src, dst, eaA, w1B[3], P[19], w2B[3], P[21], agg, ntiles);
  node_finalize_kernel<32, 16, false, false>
      <<<(NN * LAT_F + 255) / 256, 256, 0, stream>>>(h2, agg, cnt, P[22],
                                                     P[23], lv, NN);
}
